// GCN2_HBP_23055384445768
// MI455X (gfx1250) — matrix operand order repaired and checked by reading, not compiled, not run on hardware
//
#include <hip/hip_runtime.h>
#include <math.h>

#define N_NODES 20000
#define F_INN   512
#define HDIM    64
#define L_LAYERS 8
#define E_EDGES 640000
#define C_OUTD  40
#define ALPHA_C 0.1f
#define THETA_C 0.5f
#define PROJ_EPS 4e-3f
#define MIN_NORM 1e-15f

typedef __attribute__((ext_vector_type(16))) __bf16 v16bf;
typedef __attribute__((ext_vector_type(8)))  __bf16 v8bf;
typedef __attribute__((ext_vector_type(4)))  __bf16 v4bf;
typedef __attribute__((ext_vector_type(8)))  float  v8f;

__device__ __forceinline__ __bf16 f2bf(float f) {
    union { float f; unsigned u; } a; a.f = f;
    unsigned r = (a.u + 0x7FFFu + ((a.u >> 16) & 1u)) >> 16;  // RNE
    union { unsigned short s; __bf16 b; } o; o.s = (unsigned short)r;
    return o.b;
}

__device__ __forceinline__ v8f wmma_bf16(v16bf a, v16bf b, v8f c) {
    return __builtin_amdgcn_wmma_f32_16x16x32_bf16(
        false, a, false, b, (short)0, c, false, false);
}

// A fragment: 16x32 (MxK) bf16, row-major bf16 src (global or LDS), ld in elems.
__device__ __forceinline__ v16bf load_a_frag_bf(const __bf16* src, int ld,
                                                int row0, int k0, int lane) {
    const int m  = row0 + (lane & 15);
    const int kb = (lane < 16) ? 0 : 8;
    const __bf16* p = src + (size_t)m * ld + k0 + kb;
    v8bf lo = *(const v8bf*)(p);
    v8bf hi = *(const v8bf*)(p + 16);
    return __builtin_shufflevector(lo, hi, 0,1,2,3,4,5,6,7,8,9,10,11,12,13,14,15);
}

// B fragment: 32x16 (KxN) bf16 from COLUMN-major (transposed) src: srcT[n][k].
__device__ __forceinline__ v16bf load_b_frag_bf(const __bf16* srcT, int ld,
                                                int k0, int n0, int lane) {
    const int n  = n0 + (lane & 15);
    const int kb = (lane < 16) ? 0 : 8;
    const __bf16* p = srcT + (size_t)n * ld + k0 + kb;
    v8bf lo = *(const v8bf*)(p);
    v8bf hi = *(const v8bf*)(p + 16);
    return __builtin_shufflevector(lo, hi, 0,1,2,3,4,5,6,7,8,9,10,11,12,13,14,15);
}

// ---------------- one-time bf16 conversion kernels ----------------
__global__ void __launch_bounds__(256)
cvt_x_kernel(const float* __restrict__ X, __bf16* __restrict__ Xb) {
    const size_t base = ((size_t)blockIdx.x * 256 + threadIdx.x) * 4;
    if (base >= (size_t)N_NODES * F_INN) return;
    const float4 v = *reinterpret_cast<const float4*>(X + base);
    v4bf o; o[0] = f2bf(v.x); o[1] = f2bf(v.y); o[2] = f2bf(v.z); o[3] = f2bf(v.w);
    *reinterpret_cast<v4bf*>(Xb + base) = o;
}

__global__ void __launch_bounds__(256)
cvt_w0t_kernel(const float* __restrict__ W0, __bf16* __restrict__ W0t) {
    int t = blockIdx.x * 256 + threadIdx.x;              // over 64*512
    if (t >= HDIM * F_INN) return;
    int n = t >> 9, k = t & (F_INN - 1);
    W0t[t] = f2bf(W0[(size_t)k * HDIM + n]);             // W0t[n][k]
}

__global__ void __launch_bounds__(256)
cvt_wl_kernel(const float* __restrict__ Wc, __bf16* __restrict__ Wlt) {
    int t = blockIdx.x * 256 + threadIdx.x;              // over 8*64*64
    if (t >= L_LAYERS * HDIM * HDIM) return;
    int l = t >> 12, r = t & 4095, n = r >> 6, k = r & 63;
    Wlt[t] = f2bf(Wc[(size_t)l * HDIM * HDIM + k * HDIM + n]);  // Wlt[l][n][k]
}

__global__ void __launch_bounds__(256)
cvt_w1t_kernel(const float* __restrict__ W1, __bf16* __restrict__ W1tb) {
    int t = blockIdx.x * 256 + threadIdx.x;              // over 40*64*64
    if (t >= C_OUTD * HDIM * HDIM) return;
    int k = t / (HDIM * HDIM), r = t % (HDIM * HDIM);
    int j = r >> 6, i = r & 63;
    W1tb[t] = f2bf(W1[((size_t)i * HDIM + j) * C_OUTD + k]);    // W1tb[k][j][i]
}

// ------- h0 = relu(X @ W0 + b0); h = h0. One wave per 16-row strip, -------
// ------- 4 column tiles per wave so each A fragment feeds 4 WMMAs.  -------
__global__ void __launch_bounds__(32)
lin0_kernel(const __bf16* __restrict__ Xb, const __bf16* __restrict__ W0t,
            const float* __restrict__ bias, float* __restrict__ h0,
            float* __restrict__ h) {
    const int lane = threadIdx.x & 31;
    const int row0 = blockIdx.x * 16;
    v8f acc0 = {}, acc1 = {}, acc2 = {}, acc3 = {};
#pragma unroll 2
    for (int k0 = 0; k0 < F_INN; k0 += 32) {
        v16bf a  = load_a_frag_bf(Xb, F_INN, row0, k0, lane);
        v16bf b0 = load_b_frag_bf(W0t, F_INN, k0,  0, lane);
        v16bf b1 = load_b_frag_bf(W0t, F_INN, k0, 16, lane);
        v16bf b2 = load_b_frag_bf(W0t, F_INN, k0, 32, lane);
        v16bf b3 = load_b_frag_bf(W0t, F_INN, k0, 48, lane);
        acc0 = wmma_bf16(a, b0, acc0);
        acc1 = wmma_bf16(a, b1, acc1);
        acc2 = wmma_bf16(a, b2, acc2);
        acc3 = wmma_bf16(a, b3, acc3);
    }
    const int nb = lane & 15;
    const float bn0 = bias[nb], bn1 = bias[16 + nb];
    const float bn2 = bias[32 + nb], bn3 = bias[48 + nb];
#pragma unroll
    for (int r = 0; r < 8; ++r) {
        const int m = row0 + ((lane < 16) ? r : 8 + r);
        float* p0 = h0 + (size_t)m * HDIM;
        float* p1 = h  + (size_t)m * HDIM;
        float v0 = fmaxf(acc0[r] + bn0, 0.f);
        float v1 = fmaxf(acc1[r] + bn1, 0.f);
        float v2 = fmaxf(acc2[r] + bn2, 0.f);
        float v3 = fmaxf(acc3[r] + bn3, 0.f);
        p0[nb] = v0; p0[16 + nb] = v1; p0[32 + nb] = v2; p0[48 + nb] = v3;
        p1[nb] = v0; p1[16 + nb] = v1; p1[32 + nb] = v2; p1[48 + nb] = v3;
    }
}

// ---------------- CSR build: histogram -> scan -> scatter ----------------
__global__ void __launch_bounds__(256)
hist_kernel(const int* __restrict__ row, int* __restrict__ deg) {
    int t = blockIdx.x * 256 + threadIdx.x;
    if (t < E_EDGES) atomicAdd(&deg[row[t]], 1);
}

#define SCAN_CHUNK ((N_NODES + 255) / 256)
__global__ void __launch_bounds__(256)
scan_kernel(const int* __restrict__ deg, int* __restrict__ off) {
    __shared__ int part[256];
    const int tid = threadIdx.x;
    int s = 0;
    for (int i = 0; i < SCAN_CHUNK; ++i) {
        int idx = tid * SCAN_CHUNK + i;
        if (idx < N_NODES) s += deg[idx];
    }
    part[tid] = s;
    __syncthreads();
    if (tid == 0) {
        int running = 0;
        for (int j = 0; j < 256; ++j) { int v = part[j]; part[j] = running; running += v; }
        off[N_NODES] = running;
    }
    __syncthreads();
    int run = part[tid];
    for (int i = 0; i < SCAN_CHUNK; ++i) {
        int idx = tid * SCAN_CHUNK + i;
        if (idx < N_NODES) { off[idx] = run; run += deg[idx]; }
    }
}

__global__ void __launch_bounds__(256)
scatter_kernel(const int* __restrict__ row, const int* __restrict__ col,
               const float* __restrict__ ew, int* __restrict__ pos,
               int* __restrict__ sc, float* __restrict__ sw) {
    int t = blockIdx.x * 256 + threadIdx.x;
    if (t >= E_EDGES) return;
    int p = atomicAdd(&pos[row[t]], 1);
    sc[p] = col[t];
    sw[p] = ew[t];
}

// ------------- SpMM (CSR, atomic-free): ah[r] = sum_e w_e * h[col_e] -------------
__global__ void __launch_bounds__(256)
spmm_csr_kernel(const int* __restrict__ off, const int* __restrict__ sc,
                const float* __restrict__ sw, const float* __restrict__ h,
                float* __restrict__ ah) {
    const int t = blockIdx.x * 256 + threadIdx.x;
    const int r = t >> 4;                 // 16 lanes per row
    const int f = (t & 15) << 2;          // 4 features per lane
    const int start = off[r], end = off[r + 1];
    float4 acc = {0.f, 0.f, 0.f, 0.f};
    for (int e = start; e < end; ++e) {
        const int c = sc[e];
        const float w = sw[e];
        const float4 v = *reinterpret_cast<const float4*>(h + (size_t)c * HDIM + f);
        acc.x += w * v.x; acc.y += w * v.y; acc.z += w * v.z; acc.w += w * v.w;
    }
    *reinterpret_cast<float4*>(ah + (size_t)r * HDIM + f) = acc;
}

// ------- h_next = relu((1-b)*out + b*(out @ Wl)), out = .9*ah + .1*h0 -------
__global__ void __launch_bounds__(128)
layer_kernel(const float* __restrict__ ah, const float* __restrict__ h0,
             const __bf16* __restrict__ WlT, float* __restrict__ hnext,
             float beta) {
    __shared__ float out_s[16][HDIM];
    __shared__ __align__(32) __bf16 out_bf[16][72];   // 144B rows: 16B-aligned, conflict-free
    const int row0 = blockIdx.x * 16;
    for (int i = threadIdx.x; i < 16 * HDIM; i += 128) {
        int r = i >> 6, cix = i & 63;
        size_t g = (size_t)(row0 + r) * HDIM + cix;
        float v = (1.0f - ALPHA_C) * ah[g] + ALPHA_C * h0[g];
        out_s[r][cix] = v;
        out_bf[r][cix] = f2bf(v);
    }
    __syncthreads();
    const int wave = threadIdx.x >> 5, lane = threadIdx.x & 31;
    const int col0 = wave * 16;
    v8f acc = {};
#pragma unroll
    for (int k0 = 0; k0 < HDIM; k0 += 32) {
        v16bf a = load_a_frag_bf(&out_bf[0][0], 72, 0, k0, lane);
        v16bf b = load_b_frag_bf(WlT, HDIM, k0, col0, lane);
        acc = wmma_bf16(a, b, acc);
    }
    const int nl = col0 + (lane & 15);
#pragma unroll
    for (int r = 0; r < 8; ++r) {
        int ml = (lane < 16) ? r : 8 + r;
        float v = (1.0f - beta) * out_s[ml][nl] + beta * acc[r];
        v = v > 0.f ? v : 0.f;
        hnext[(size_t)(row0 + ml) * HDIM + nl] = v;
    }
}

// ------- head: g_k = h^T B_k h, then Poincare maps + log_softmax -------
__global__ void __launch_bounds__(128)
head_kernel(const float* __restrict__ h, const __bf16* __restrict__ W1tb,
            const float* __restrict__ b1, float* __restrict__ out) {
    __shared__ float Hs[16][HDIM + 1];
    __shared__ __align__(32) __bf16 Hbf[16][72];
    __shared__ float Gs[16][C_OUTD];
    const int row0 = blockIdx.x * 16;
    for (int i = threadIdx.x; i < 16 * HDIM; i += 128) {
        int r = i >> 6, cix = i & 63;
        float v = h[(size_t)(row0 + r) * HDIM + cix];
        Hs[r][cix] = v;
        Hbf[r][cix] = f2bf(v);
    }
    for (int i = threadIdx.x; i < 16 * C_OUTD; i += 128)
        Gs[i / C_OUTD][i % C_OUTD] = 0.f;
    __syncthreads();

    const int wave = threadIdx.x >> 5, lane = threadIdx.x & 31;
    const int col0 = wave * 16;
    const v16bf a0 = load_a_frag_bf(&Hbf[0][0], 72, 0, 0, lane);
    const v16bf a1 = load_a_frag_bf(&Hbf[0][0], 72, 0, 32, lane);
    const int nl = col0 + (lane & 15);

    // hoist loop-invariant H[m, nl] values out of the class loop
    float hrow[8];
#pragma unroll
    for (int r = 0; r < 8; ++r) {
        int ml = (lane < 16) ? r : 8 + r;
        hrow[r] = Hs[ml][nl];
    }

    for (int k = 0; k < C_OUTD; ++k) {
        const __bf16* BkT = W1tb + (size_t)k * HDIM * HDIM;  // column-major B_k
        v8f acc = {};
        acc = wmma_bf16(a0, load_b_frag_bf(BkT, HDIM, 0, col0, lane), acc);
        acc = wmma_bf16(a1, load_b_frag_bf(BkT, HDIM, 32, col0, lane), acc);
#pragma unroll
        for (int r = 0; r < 8; ++r) {
            int ml = (lane < 16) ? r : 8 + r;
            float p = acc[r] * hrow[r];             // U[m,j] * H[m,j]
            p += __shfl_xor(p, 8, 32);
            p += __shfl_xor(p, 4, 32);
            p += __shfl_xor(p, 2, 32);
            p += __shfl_xor(p, 1, 32);
            if ((lane & 15) == 0) atomicAdd(&Gs[ml][k], p);   // ds_add_f32
        }
    }
    __syncthreads();

    if (threadIdx.x < 16) {
        const int node = row0 + threadIdx.x;
        float s = 0.f;
        for (int j = 0; j < HDIM; ++j) { float v = Hs[threadIdx.x][j]; s += v * v; }
        const float maxnorm = 1.0f - PROJ_EPS;                // sqrt(c)=1
        float pn = fmaxf(s, MIN_NORM);                        // ||outer(h,h)||_F
        float f1 = (pn > maxnorm) ? (maxnorm / pn) : 1.0f;    // proj
        float pn2 = fmaxf(fminf(pn, maxnorm), MIN_NORM);
        float f2 = atanhf(fminf(pn2, 1.0f - 1e-7f)) / pn2;    // logmap0
        float scale = f1 * f2;
        float y[C_OUTD];
        float sum2 = 0.f;
        for (int k = 0; k < C_OUTD; ++k) {
            float v = scale * Gs[threadIdx.x][k] + b1[k];
            y[k] = v; sum2 += v * v;
        }
        float un = fmaxf(sqrtf(sum2), MIN_NORM);
        float ef = tanhf(un) / un;                            // expmap0
        float zn = fmaxf(tanhf(un), MIN_NORM);
        float f3 = (zn > maxnorm) ? (maxnorm / zn) : 1.0f;    // proj
        float sc = ef * f3;
        float mx = -1e30f;
        for (int k = 0; k < C_OUTD; ++k) { y[k] *= sc; mx = fmaxf(mx, y[k]); }
        float se = 0.f;
        for (int k = 0; k < C_OUTD; ++k) se += expf(y[k] - mx);
        float lse = mx + logf(se);
        for (int k = 0; k < C_OUTD; ++k)
            out[(size_t)node * C_OUTD + k] = y[k] - lse;
    }
}

static inline size_t align256(size_t x) { return (x + 255) & ~(size_t)255; }

extern "C" void kernel_launch(void* const* d_in, const int* in_sizes, int n_in,
                              void* d_out, int out_size, void* d_ws, size_t ws_size,
                              hipStream_t stream) {
    (void)in_sizes; (void)n_in; (void)out_size; (void)ws_size;
    const float* x     = (const float*)d_in[0];
    const int*   row   = (const int*)  d_in[1];
    const int*   col   = (const int*)  d_in[2];
    const float* ew    = (const float*)d_in[3];
    const float* w0    = (const float*)d_in[4];
    const float* b0    = (const float*)d_in[5];
    const float* convw = (const float*)d_in[6];
    const float* w1    = (const float*)d_in[7];
    const float* b1    = (const float*)d_in[8];
    float* out = (float*)d_out;

    char* p = (char*)d_ws;
    size_t o = 0;
    float* h0f = (float*)(p + o); o = align256(o + (size_t)N_NODES * HDIM * 4);
    float* haf = (float*)(p + o); o = align256(o + (size_t)N_NODES * HDIM * 4);
    float* hbf = (float*)(p + o); o = align256(o + (size_t)N_NODES * HDIM * 4);
    float* ah  = (float*)(p + o); o = align256(o + (size_t)N_NODES * HDIM * 4);
    __bf16* xbf  = (__bf16*)(p + o); o = align256(o + (size_t)N_NODES * F_INN * 2);
    __bf16* w0t  = (__bf16*)(p + o); o = align256(o + (size_t)HDIM * F_INN * 2);
    __bf16* wlt  = (__bf16*)(p + o); o = align256(o + (size_t)L_LAYERS * HDIM * HDIM * 2);
    __bf16* w1tb = (__bf16*)(p + o); o = align256(o + (size_t)C_OUTD * HDIM * HDIM * 2);
    int*   deg = (int*)(p + o); o = align256(o + (size_t)N_NODES * 4);
    int*   off = (int*)(p + o); o = align256(o + (size_t)(N_NODES + 1) * 4);
    int*   pos = (int*)(p + o); o = align256(o + (size_t)N_NODES * 4);
    int*   sc  = (int*)(p + o); o = align256(o + (size_t)E_EDGES * 4);
    float* sw  = (float*)(p + o); o = align256(o + (size_t)E_EDGES * 4);

    // one-time conversions + lin0
    cvt_x_kernel  <<<((size_t)N_NODES * F_INN / 4 + 255) / 256, 256, 0, stream>>>(x, xbf);
    cvt_w0t_kernel<<<(HDIM * F_INN + 255) / 256, 256, 0, stream>>>(w0, w0t);
    cvt_wl_kernel <<<(L_LAYERS * HDIM * HDIM + 255) / 256, 256, 0, stream>>>(convw, wlt);
    cvt_w1t_kernel<<<(C_OUTD * HDIM * HDIM + 255) / 256, 256, 0, stream>>>(w1, w1tb);

    lin0_kernel<<<N_NODES / 16, 32, 0, stream>>>(xbf, w0t, b0, h0f, haf);

    // CSR build (int atomics only)
    hipMemsetAsync(deg, 0, (size_t)N_NODES * 4, stream);
    hist_kernel   <<<(E_EDGES + 255) / 256, 256, 0, stream>>>(row, deg);
    scan_kernel   <<<1, 256, 0, stream>>>(deg, off);
    hipMemcpyAsync(pos, off, (size_t)N_NODES * 4, hipMemcpyDeviceToDevice, stream);
    scatter_kernel<<<(E_EDGES + 255) / 256, 256, 0, stream>>>(row, col, ew, pos, sc, sw);

    const dim3 gemm_grid(N_NODES / 16);
    float* hc = haf; float* hn = hbf;
    for (int l = 0; l < L_LAYERS; ++l) {
        spmm_csr_kernel<<<(N_NODES * 16) / 256, 256, 0, stream>>>(off, sc, sw, hc, ah);
        float beta = logf(THETA_C / (float)(l + 1) + 1.0f);
        layer_kernel<<<gemm_grid, 128, 0, stream>>>(
            ah, h0f, wlt + (size_t)l * HDIM * HDIM, hn, beta);
        float* t = hc; hc = hn; hn = t;
    }
    head_kernel<<<gemm_grid, 128, 0, stream>>>(hc, w1tb, b1, out);
}
